// IOU_SSIM_9002251453025
// MI455X (gfx1250) — matrix operand order repaired and checked by reading, not compiled, not run on hardware
//
#include <hip/hip_runtime.h>
#include <hip/hip_bf16.h>

typedef __attribute__((ext_vector_type(16))) _Float16 v16h;
typedef __attribute__((ext_vector_type(8)))  float    v8f;

#define BB   4
#define HH   640
#define WW   640
#define NG   16
#define NP   64
#define NPAIR (BB*NG*NP)          // 4096
#define NCROP (BB*(NG+NP))        // 320
#define C1N  (0.01f*0.01f)        // C1 rescaled to [0,1] pixel domain
#define C2N  (0.03f*0.03f)
#define THRESH 0.3f

__device__ __forceinline__ int laneId() { return threadIdx.x & 31; }

__device__ __forceinline__ void gaussWeights(float* gw) {
  float s = 0.f;
#pragma unroll
  for (int t = 0; t < 11; ++t) { float x = (float)t - 5.0f; float v = expf(-x*x/(2.0f*1.5f*1.5f)); gw[t] = v; s += v; }
#pragma unroll
  for (int t = 0; t < 11; ++t) gw[t] /= s;
}

__device__ __forceinline__ float kband(const float* gw, int i, int j) {
  int d = j - i;
  return (d >= -5 && d <= 5) ? gw[d + 5] : 0.0f;
}

// ---- WMMA operand builders (wave32, 16x16x32 f16 -> f32) ----
// A (16x32 f16): lane<16 holds row r=lane, elems 0..7 -> K0..7, 8..15 -> K16..23;
//                lane>=16 same row r=lane-16, elems -> K8..15 / K24..31.
__device__ __forceinline__ v16h buildA_f32src(const float* src, int rb) {
  int l = laneId(); int r = rb + (l & 15); int h = l >> 4;
  v16h a;
#pragma unroll
  for (int e = 0; e < 16; ++e) {
    int kk = (e < 8) ? (h*8 + e) : (16 + h*8 + (e - 8));
    a[e] = (_Float16)src[r*32 + kk];
  }
  return a;
}
__device__ __forceinline__ v16h buildA_K(const float* gw, int rb) {
  int l = laneId(); int r = rb + (l & 15); int h = l >> 4;
  v16h a;
#pragma unroll
  for (int e = 0; e < 16; ++e) {
    int kk = (e < 8) ? (h*8 + e) : (16 + h*8 + (e - 8));
    a[e] = (_Float16)kband(gw, r, kk);
  }
  return a;
}
// B (32x16 f16): lane holds column n=lane&15; lane<16 -> K0..7/K16..23, lane>=16 -> K8..15/K24..31.
__device__ __forceinline__ v16h buildB_f32src(const float* src, int cb) {
  int l = laneId(); int n = cb + (l & 15); int h = l >> 4;
  v16h b;
#pragma unroll
  for (int e = 0; e < 16; ++e) { int kk = (e < 8) ? (h*8 + e) : (16 + h*8 + (e - 8)); b[e] = (_Float16)src[kk*32 + n]; }
  return b;
}
__device__ __forceinline__ v16h buildB_prod(const float* s1, const float* s2, int cb) {
  int l = laneId(); int n = cb + (l & 15); int h = l >> 4;
  v16h b;
#pragma unroll
  for (int e = 0; e < 16; ++e) { int kk = (e < 8) ? (h*8 + e) : (16 + h*8 + (e - 8)); int idx = kk*32 + n; b[e] = (_Float16)(s1[idx]*s2[idx]); }
  return b;
}
__device__ __forceinline__ v16h buildB_K(const float* gw, int cb) {
  int l = laneId(); int n = cb + (l & 15); int h = l >> 4;
  v16h b;
#pragma unroll
  for (int e = 0; e < 16; ++e) { int kk = (e < 8) ? (h*8 + e) : (16 + h*8 + (e - 8)); b[e] = (_Float16)kband(gw, kk, n); }
  return b;
}
// D/C (16x16 f32): vgpr j holds row m = mb + j + 8*(lane>=16), col n = nb + (lane&15)
__device__ __forceinline__ void storeTile(float* dst, v8f acc, int mb, int nb) {
  int l = laneId(); int n = nb + (l & 15); int mh = l >> 4;
#pragma unroll
  for (int j = 0; j < 8; ++j) dst[(mb + j + 8*mh)*32 + n] = acc[j];
}

__device__ __forceinline__ v8f wmma16(v16h a, v16h b, v8f c) {
  return __builtin_amdgcn_wmma_f32_16x16x32_f16(false, a, false, b, (short)0, c, false, false);
}

// Y = K * S * K   (S = sA .* sB if sB != nullptr else sA), via 8 WMMAs + LDS turnaround.
__device__ __forceinline__ void conv32(v16h KA0, v16h KA1, v16h KB0, v16h KB1,
                                       const float* sA, const float* sB,
                                       float* ldsT, v8f acc[4]) {
  v16h B0, B1;
  if (sB) { B0 = buildB_prod(sA, sB, 0); B1 = buildB_prod(sA, sB, 16); }
  else    { B0 = buildB_f32src(sA, 0);   B1 = buildB_f32src(sA, 16);  }
  v8f z = {};
  v8f t00 = wmma16(KA0, B0, z);
  v8f t01 = wmma16(KA0, B1, z);
  v8f t10 = wmma16(KA1, B0, z);
  v8f t11 = wmma16(KA1, B1, z);
  storeTile(ldsT, t00, 0, 0);  storeTile(ldsT, t01, 0, 16);
  storeTile(ldsT, t10, 16, 0); storeTile(ldsT, t11, 16, 16);
  asm volatile("s_wait_dscnt 0" ::: "memory");   // same-wave LDS RAW turnaround
  v16h TA0 = buildA_f32src(ldsT, 0);
  v16h TA1 = buildA_f32src(ldsT, 16);
  acc[0] = wmma16(TA0, KB0, z);
  acc[1] = wmma16(TA0, KB1, z);
  acc[2] = wmma16(TA1, KB0, z);
  acc[3] = wmma16(TA1, KB1, z);
}

// ---------- Kernel 1: crop-resize + per-crop mu / E[x^2] via WMMA convs ----------
__global__ void __launch_bounds__(128)
crop_stats_kernel(const float* __restrict__ imgs,
                  const float* __restrict__ gt_boxes,
                  const float* __restrict__ pred_boxes,
                  float* __restrict__ crops, float* __restrict__ mu, float* __restrict__ m2) {
  __shared__ float sCrop[3*1024];
  __shared__ float sT[4][1024];
  __shared__ float sFX[32], sFY[32];
  __shared__ int   sIX0[32], sIX1[32], sIY0[32], sIY1[32];

  int crop = blockIdx.x;
  int b; const float* box;
  if (crop < BB*NG) { b = crop / NG; box = gt_boxes + (size_t)crop*4; }
  else { int c2 = crop - BB*NG; b = c2 / NP; box = pred_boxes + (size_t)c2*4; }

  float cx = box[0], cy = box[1], w = box[2], h = box[3];
  float x0 = fminf(fmaxf(floorf(cx - w*0.5f), 0.f), (float)WW - 1.f);
  float x1 = fminf(fmaxf(floorf(cx + w*0.5f), 0.f), (float)WW);
  float y0 = fminf(fmaxf(floorf(cy - h*0.5f), 0.f), (float)HH - 1.f);
  float y1 = fminf(fmaxf(floorf(cy + h*0.5f), 0.f), (float)HH);
  float wp = fmaxf(x1 - x0, 1.f), hp = fmaxf(y1 - y0, 1.f);

  int t = threadIdx.x;
  if (t < 32) {
    float d = (float)t + 0.5f;
    float sx = x0 + fminf(fmaxf(d*wp*(1.0f/32.f) - 0.5f, 0.f), wp - 1.f);
    float fx0 = floorf(sx);
    sFX[t] = sx - fx0;
    int ix0 = (int)fx0;
    sIX0[t] = ix0;
    sIX1[t] = min(ix0 + 1, (int)(x0 + wp - 1.f));
  } else if (t < 64) {
    int tt = t - 32;
    float d = (float)tt + 0.5f;
    float sy = y0 + fminf(fmaxf(d*hp*(1.0f/32.f) - 0.5f, 0.f), hp - 1.f);
    float fy0 = floorf(sy);
    sFY[tt] = sy - fy0;
    int iy0 = (int)fy0;
    sIY0[tt] = iy0;
    sIY1[tt] = min(iy0 + 1, (int)(y0 + hp - 1.f));
  }
  __syncthreads();

  const float* img = imgs + (size_t)b*3*HH*WW;
  for (int idx = t; idx < 3*1024; idx += 128) {
    int c = idx >> 10; int pix = idx & 1023; int y = pix >> 5; int x = pix & 31;
    const float* ch = img + (size_t)c*HH*WW;
    int ix0 = sIX0[x], ix1 = sIX1[x], iy0 = sIY0[y], iy1 = sIY1[y];
    float fx = sFX[x], fy = sFY[y];
    float v = ch[iy0*WW+ix0]*(1.f-fy)*(1.f-fx) + ch[iy0*WW+ix1]*(1.f-fy)*fx
            + ch[iy1*WW+ix0]*fy*(1.f-fx)       + ch[iy1*WW+ix1]*fy*fx;
    v *= (1.0f/255.0f);                 // normalized domain (SSIM scale-invariant)
    sCrop[idx] = v;
    crops[(size_t)crop*3072 + idx] = v;
  }
  __syncthreads();

  float gw[11]; gaussWeights(gw);
  v16h KA0 = buildA_K(gw, 0), KA1 = buildA_K(gw, 16);
  v16h KB0 = buildB_K(gw, 0), KB1 = buildB_K(gw, 16);

  int wave = t >> 5;
  for (int task = wave; task < 6; task += 4) {   // 3 channels x {x, x^2}
    int c = task >> 1; int sq = task & 1;
    const float* src = sCrop + c*1024;
    v8f acc[4];
    conv32(KA0, KA1, KB0, KB1, src, sq ? src : nullptr, sT[wave], acc);
    float* dst = (sq ? m2 : mu) + (size_t)crop*3072 + c*1024;
    storeTile(dst, acc[0], 0, 0);  storeTile(dst, acc[1], 0, 16);
    storeTile(dst, acc[2], 16, 0); storeTile(dst, acc[3], 16, 16);
  }
}

// ---------- Kernel 2: per (pair, channel): conv(g.*p) via WMMA, SSIM + L1 partials ----------
__global__ void __launch_bounds__(128)
pair_kernel(const float* __restrict__ gt_boxes, const float* __restrict__ pred_boxes,
            const float* __restrict__ crops, const float* __restrict__ mu, const float* __restrict__ m2,
            float* __restrict__ l1_part, float* __restrict__ ssim_part, float* __restrict__ wv) {
  __shared__ float sG[4][1024], sP[4][1024], sT[4][1024];
  int wave = threadIdx.x >> 5; int l = threadIdx.x & 31;
  int task = blockIdx.x*4 + wave;           // 0..12287, exact fill
  int pair = task / 3; int c = task % 3;
  int b = pair / (NG*NP); int rem = pair % (NG*NP); int i = rem / NP; int j = rem % NP;
  int gcrop = b*NG + i;
  int pcrop = BB*NG + b*NP + j;

  const float* gsrc = crops + (size_t)gcrop*3072 + c*1024;
  const float* psrc = crops + (size_t)pcrop*3072 + c*1024;
  __builtin_prefetch(gsrc, 0, 0);
  __builtin_prefetch(psrc, 0, 0);

  float* myG = sG[wave]; float* myP = sP[wave];
  float l1s = 0.f;
#pragma unroll 4
  for (int e = 0; e < 32; ++e) {
    int idx = e*32 + l;
    float gv = gsrc[idx], pv = psrc[idx];
    myG[idx] = gv; myP[idx] = pv;
    l1s += fabsf(gv - pv);                  // == |g-p|/255 in raw domain
  }
  asm volatile("s_wait_dscnt 0" ::: "memory");

  float gw[11]; gaussWeights(gw);
  v16h KA0 = buildA_K(gw, 0), KA1 = buildA_K(gw, 16);
  v16h KB0 = buildB_K(gw, 0), KB1 = buildB_K(gw, 16);
  v8f acc[4];
  conv32(KA0, KA1, KB0, KB1, myG, myP, sT[wave], acc);   // conv(g.*p)

  const float* mug = mu + (size_t)gcrop*3072 + c*1024;
  const float* m2g = m2 + (size_t)gcrop*3072 + c*1024;
  const float* mup = mu + (size_t)pcrop*3072 + c*1024;
  const float* m2p = m2 + (size_t)pcrop*3072 + c*1024;

  float ssims = 0.f;
  int nbase = l & 15; int mh = l >> 4;
#pragma unroll
  for (int tile = 0; tile < 4; ++tile) {
    int mb = (tile >> 1)*16, nb = (tile & 1)*16;
    int n = nb + nbase;
#pragma unroll
    for (int jj = 0; jj < 8; ++jj) {
      int m = mb + jj + 8*mh;
      int idx = m*32 + n;
      float a = mug[idx], p = mup[idx];
      float sg  = m2g[idx] - a*a;
      float sp  = m2p[idx] - p*p;
      float sgp = acc[tile][jj] - a*p;
      float num = (2.f*a*p + C1N) * (2.f*sgp + C2N);
      float den = (a*a + p*p + C1N) * (sg + sp + C2N);
      ssims += num / den;
    }
  }
#pragma unroll
  for (int mk = 16; mk >= 1; mk >>= 1) {
    l1s  += __shfl_xor(l1s,  mk, 32);
    ssims += __shfl_xor(ssims, mk, 32);
  }
  if (l == 0) {
    l1_part[task]   = l1s;
    ssim_part[task] = ssims;
    if (c == 0) {
      const float* gbx = gt_boxes   + (size_t)(b*NG + i)*4;
      const float* pbx = pred_boxes + (size_t)(b*NP + j)*4;
      float gx=gbx[0], gy=gbx[1], gwd=gbx[2], ght=gbx[3];
      float px=pbx[0], py=pbx[1], pwd=pbx[2], pht=pbx[3];
      float tlx = fmaxf(gx - gwd*0.5f, px - pwd*0.5f);
      float tly = fmaxf(gy - ght*0.5f, py - pht*0.5f);
      float brx = fminf(gx + gwd*0.5f, px + pwd*0.5f);
      float bry = fminf(gy + ght*0.5f, py + pht*0.5f);
      float ag = gwd*ght, ap = pwd*pht;
      float en = (tlx < brx && tly < bry) ? 1.f : 0.f;
      float ai = (brx - tlx)*(bry - tly)*en;
      float iou = ai / (ag + ap - ai + 1e-16f);
      wv[pair] = (iou > THRESH && pwd > 2.f && pht > 2.f) ? 1.f : 0.f;
    }
  }
}

// ---------- Kernel 3: deterministic fixed-order reduction ----------
__global__ void __launch_bounds__(256)
finalize_kernel(const float* __restrict__ l1_part, const float* __restrict__ ssim_part,
                const float* __restrict__ wv, float* __restrict__ out) {
  __shared__ float sC[256], sL[256], sS[256];
  int t = threadIdx.x;
  float cnt = 0.f, sl = 0.f, ss = 0.f;
  for (int pair = t; pair < NPAIR; pair += 256) {
    float w = wv[pair];
    cnt += w;
    float lsum = l1_part[pair*3] + l1_part[pair*3+1] + l1_part[pair*3+2];
    float ssum = ssim_part[pair*3] + ssim_part[pair*3+1] + ssim_part[pair*3+2];
    sl += w * lsum;
    ss += w * ssum;
  }
  sC[t] = cnt; sL[t] = sl; sS[t] = ss;
  __syncthreads();
  for (int s = 128; s > 0; s >>= 1) {
    if (t < s) { sC[t] += sC[t+s]; sL[t] += sL[t+s]; sS[t] += sS[t+s]; }
    __syncthreads();
  }
  if (t == 0) {
    float c = sC[0];
    float denom = fmaxf(c, 1.f);
    float meanL1 = (sL[0] * (1.0f/3072.f)) / denom;   // mean over 3*32*32
    float meanSS = (sS[0] * (1.0f/3072.f)) / denom;
    float loss = meanL1 + (1.f - meanSS);
    out[0] = (c > 0.f) ? loss : 0.f;
  }
}

extern "C" void kernel_launch(void* const* d_in, const int* in_sizes, int n_in,
                              void* d_out, int out_size, void* d_ws, size_t ws_size,
                              hipStream_t stream) {
  (void)in_sizes; (void)n_in; (void)out_size; (void)ws_size;
  const float* imgs = (const float*)d_in[0];
  const float* gt   = (const float*)d_in[1];
  const float* pr   = (const float*)d_in[2];

  float* ws    = (float*)d_ws;
  float* crops = ws;                          // 320*3072
  float* mu    = crops + (size_t)NCROP*3072;  // 320*3072
  float* m2    = mu    + (size_t)NCROP*3072;  // 320*3072
  float* l1p   = m2    + (size_t)NCROP*3072;  // 12288
  float* ssp   = l1p   + (size_t)NPAIR*3;     // 12288
  float* wvv   = ssp   + (size_t)NPAIR*3;     // 4096

  crop_stats_kernel<<<NCROP, 128, 0, stream>>>(imgs, gt, pr, crops, mu, m2);
  pair_kernel<<<NPAIR*3/4, 128, 0, stream>>>(gt, pr, crops, mu, m2, l1p, ssp, wvv);
  finalize_kernel<<<1, 256, 0, stream>>>(l1p, ssp, wvv, (float*)d_out);
}
